// MH_SelfAttentionNarrow_73194832658892
// MI455X (gfx1250) — matrix operand order repaired and checked by reading, not compiled, not run on hardware
//
#include <hip/hip_runtime.h>

#define B_  2
#define S_  2048
#define D_  1024
#define H_  16
#define HD_ 64
#define SSTR 2052   // attn LDS row stride (floats); 16B aligned, avoids half-wave bank overlap
#define LDSTR 68    // qkv staging stride (floats)

typedef __attribute__((ext_vector_type(16))) __bf16         v16bf;
typedef __attribute__((ext_vector_type(16))) unsigned short v16u;
typedef __attribute__((ext_vector_type(8)))  float          v8f;

union frag_u { v16u v; v16bf bf; uint4 q[2]; };

// hardware RTNE conversion: lowers to v_cvt(_pk)_bf16_f32 instead of integer rounding
__device__ __forceinline__ unsigned short f2bfu(float f) {
  return __builtin_bit_cast(unsigned short, (__bf16)f);
}

__device__ __forceinline__ v8f v8f_zero() {
  v8f z = {0.f, 0.f, 0.f, 0.f, 0.f, 0.f, 0.f, 0.f};
  return z;
}

__device__ __forceinline__ v8f wmma_bf16(v16bf a, v16bf b, v8f c) {
  return __builtin_amdgcn_wmma_f32_16x16x32_bf16(false, a, false, b, (short)0, c, false, false);
}

// ---- wide fragment loads ----

// A 16x32 fragment from row-major bf16: lane holds one contiguous 32B run
__device__ __forceinline__ v16bf ld_a_u16(const unsigned short* s, int ld, int k0) {
  const int lane = threadIdx.x & 31, r = lane & 15, hs = lane >> 4;
  const unsigned short* p = s + (size_t)r * ld + k0 + 16 * hs;
  frag_u f;
  f.q[0] = *(const uint4*)(p);        // K = k0+16hs .. +7
  f.q[1] = *(const uint4*)(p + 8);    // K = k0+16hs+8 .. +15
  return f.bf;
}

// A 16x32 fragment from row-major f32 (4x b128 loads + packed cvt)
__device__ __forceinline__ v16bf ld_a_f32(const float* s, int ld, int k0) {
  const int lane = threadIdx.x & 31, r = lane & 15, hs = lane >> 4;
  const float* p = s + (size_t)r * ld + k0 + 16 * hs;
  const float4 a = *(const float4*)(p);
  const float4 b = *(const float4*)(p + 4);
  const float4 c = *(const float4*)(p + 8);
  const float4 d = *(const float4*)(p + 12);
  v16bf r16;
  r16[0]  = (__bf16)a.x; r16[1]  = (__bf16)a.y; r16[2]  = (__bf16)a.z; r16[3]  = (__bf16)a.w;
  r16[4]  = (__bf16)b.x; r16[5]  = (__bf16)b.y; r16[6]  = (__bf16)b.z; r16[7]  = (__bf16)b.w;
  r16[8]  = (__bf16)c.x; r16[9]  = (__bf16)c.y; r16[10] = (__bf16)c.z; r16[11] = (__bf16)c.w;
  r16[12] = (__bf16)d.x; r16[13] = (__bf16)d.y; r16[14] = (__bf16)d.z; r16[15] = (__bf16)d.w;
  return r16;
}

// B 32x16 fragment from a pre-swizzled 1KB block: [lane][16 bf16] -> 2x b128 per lane
__device__ __forceinline__ v16bf ld_b_swz(const unsigned short* blk) {
  const int lane = threadIdx.x & 31;
  const unsigned short* p = blk + lane * 16;
  frag_u f;
  f.q[0] = *(const uint4*)(p);
  f.q[1] = *(const uint4*)(p + 8);
  return f.bf;
}

__device__ __forceinline__ void store32B(unsigned short* dp, const v16bf& raw) {
  frag_u f; f.bf = raw;
  *(uint4*)(dp)     = f.q[0];
  *(uint4*)(dp + 8) = f.q[1];
}

// ---- kernel 1: weights -> bf16, swizzled into B-fragment blocks (coalesced writes) ----
// block element mapping: k_local = 32*kchunk + 16*(lane>>4) + w16 ; n = 16*ntile + (lane&15)
__global__ __launch_bounds__(256) void prep_weights(
    const float* __restrict__ Wq, const float* __restrict__ Wk,
    const float* __restrict__ Wv, const float* __restrict__ Wz,
    unsigned short* __restrict__ WqS, unsigned short* __restrict__ WkS,
    unsigned short* __restrict__ WvS, unsigned short* __restrict__ WzS) {
  const int stride = gridDim.x * blockDim.x;
  const int tid = blockIdx.x * blockDim.x + threadIdx.x;
  // per-head weights: layout [h][dchunk(2)][etile(4)][512]
  for (int i = tid; i < H_ * HD_ * HD_; i += stride) {
    const int h = i >> 12, rem = i & 4095;
    const int dchunk = rem >> 11, rem2 = rem & 2047;
    const int et = rem2 >> 9, pos = rem2 & 511;
    const int lt = pos >> 4, w16 = pos & 15;
    const int d = dchunk * 32 + ((lt >> 4) << 4) + w16;
    const int e = et * 16 + (lt & 15);
    const size_t src = (size_t)(h * HD_ + e) * HD_ + d;   // W[h][e][d]
    WqS[i] = f2bfu(Wq[src]);
    WkS[i] = f2bfu(Wk[src]);
    WvS[i] = f2bfu(Wv[src]);
  }
  // Wz: layout [dchunk(32)][etile(64)][512]
  for (int i = tid; i < D_ * D_; i += stride) {
    const int dchunk = i >> 15, rem = i & 32767;
    const int et = rem >> 9, pos = rem & 511;
    const int lt = pos >> 4, w16 = pos & 15;
    const int d = dchunk * 32 + ((lt >> 4) << 4) + w16;
    const int e = et * 16 + (lt & 15);
    WzS[i] = f2bfu(Wz[(size_t)e * D_ + d]);               // Wz[e][d]
  }
}

// ---- kernel 2: per-head projection; 8 waves cover 128 rows of one (b,h).
// mode 0: row-major bf16 out (Q, consumed as A)
// mode 1: B-swizzled out (V): blocks [keychunk S/32][ftile 4][512]
// mode 2: Bt-swizzled out (K^T): blocks [keytile S/16][fchunk 2][512]
__global__ __launch_bounds__(256) void qkv_proj(
    const float* __restrict__ X, const unsigned short* __restrict__ WS,
    unsigned short* __restrict__ dst, float scale, int mode) {
  __shared__ float tile[128][LDSTR];
  const int lane = threadIdx.x & 31;
  const int wave = threadIdx.x >> 5;
  const int tpg = (S_ / 16) / 8;                  // 16 groups per head
  const int g = blockIdx.x;
  const int s0base = (g % tpg) * 128;
  const int h = (g / tpg) % H_;
  const int b = g / (tpg * H_);
  const int s0 = s0base + wave * 16;
  const float* xs = X + ((size_t)b * S_ + s0) * D_ + h * HD_;
  const unsigned short* wsh = WS + (size_t)h * 4096;

  v8f acc[4] = {v8f_zero(), v8f_zero(), v8f_zero(), v8f_zero()};
#pragma unroll
  for (int kc = 0; kc < HD_; kc += 32) {
    const v16bf a = ld_a_f32(xs, D_, kc);
#pragma unroll
    for (int nt = 0; nt < 4; ++nt)
      acc[nt] = wmma_bf16(a, ld_b_swz(wsh + ((kc >> 5) * 4 + nt) * 512), acc[nt]);
  }
  const int col = lane & 15, hs = lane >> 4;
#pragma unroll
  for (int nt = 0; nt < 4; ++nt)
#pragma unroll
    for (int v = 0; v < 8; ++v)
      tile[wave * 16 + v + 8 * hs][nt * 16 + col] = acc[nt][v] * scale;
  __syncthreads();

  // cooperative swizzled emission: 512 slots of 32B, coalesced b128 stores
  for (int si = threadIdx.x; si < 512; si += 256) {
    v16bf raw;
    if (mode == 0) {
      const int srow = si >> 2, f0 = (si & 3) << 4;
#pragma unroll
      for (int w = 0; w < 16; ++w) raw[w] = (__bf16)tile[srow][f0 + w];
      unsigned short* dp = dst + (((size_t)b * H_ + h) * S_ + s0base + srow) * HD_ + f0;
      store32B(dp, raw);
    } else if (mode == 1) {
      const int blk = si >> 5, lt = si & 31;
      const int kchunk = blk >> 2, ft = blk & 3;
      const int n = ft * 16 + (lt & 15);
      const int kbase = kchunk * 32 + ((lt >> 4) << 4);
#pragma unroll
      for (int w = 0; w < 16; ++w) raw[w] = (__bf16)tile[kbase + w][n];
      unsigned short* dp = dst +
          ((((size_t)b * H_ + h) * (S_ / 32) + (s0base >> 5) + kchunk) * 4 + ft) * 512 + lt * 16;
      store32B(dp, raw);
    } else {
      const int blk = si >> 5, lt = si & 31;
      const int ntl = blk >> 1, kchunk = blk & 1;
      const int srow = ntl * 16 + (lt & 15);
      const int fbase = kchunk * 32 + ((lt >> 4) << 4);
#pragma unroll
      for (int w = 0; w < 16; ++w) raw[w] = (__bf16)tile[srow][fbase + w];
      unsigned short* dp = dst +
          ((((size_t)b * H_ + h) * (S_ / 16) + (s0base >> 4) + ntl) * 2 + kchunk) * 512 + lt * 16;
      store32B(dp, raw);
    }
  }
}

// ---- kernel 3: attention; 1 WG (4 waves) per (b, 16-query tile), loops heads.
// Score tile 16x2048 f32 lives in LDS (131 KB of the 320 KB WGP LDS).
__global__ __launch_bounds__(128) void attn_kernel(
    const unsigned short* __restrict__ Qb, const unsigned short* __restrict__ KS,
    const unsigned short* __restrict__ VS, unsigned short* __restrict__ Zb,
    float* __restrict__ attn_mean) {
  extern __shared__ __align__(16) float sc[];     // [16][SSTR]
  const int lane = threadIdx.x & 31;
  const int wave = threadIdx.x >> 5;
  const int b  = blockIdx.y;
  const int q0 = blockIdx.x * 16;
  const int col = lane & 15, hs = lane >> 4;

  for (int h = 0; h < H_; ++h) {
    const unsigned short* Qp = Qb + (((size_t)b * H_ + h) * S_ + q0) * HD_;
    const unsigned short* Kh = KS + ((size_t)b * H_ + h) * (S_ / 16) * 1024;   // [keytile][fchunk][512]
    const unsigned short* Vh = VS + ((size_t)b * H_ + h) * (S_ / 32) * 2048;   // [keychunk][ftile][512]

    const v16bf aq0 = ld_a_u16(Qp, HD_, 0);
    const v16bf aq1 = ld_a_u16(Qp, HD_, 32);

    // scores: each wave owns 32 key tiles (512 columns); B fragments are 2x b128
    for (int nt = wave * 32; nt < (wave + 1) * 32; ++nt) {
      const unsigned short* blk = Kh + (size_t)nt * 1024;
      __builtin_prefetch(blk + 1024, 0, 0);
      v8f acc = v8f_zero();
      acc = wmma_bf16(aq0, ld_b_swz(blk), acc);
      acc = wmma_bf16(aq1, ld_b_swz(blk + 512), acc);
      const int k0 = nt * 16;
#pragma unroll
      for (int v = 0; v < 8; ++v)
        sc[(size_t)(v + 8 * hs) * SSTR + k0 + col] = acc[v];
    }
    __syncthreads();

    // softmax: each wave owns 4 query rows; wave32 shuffle reductions
    for (int r = wave * 4; r < wave * 4 + 4; ++r) {
      float* row = sc + (size_t)r * SSTR;
      float m = -3.0e38f;
      for (int j = lane; j < S_; j += 32) m = fmaxf(m, row[j]);
#pragma unroll
      for (int o = 16; o > 0; o >>= 1) m = fmaxf(m, __shfl_xor(m, o, 32));
      float sum = 0.f;
      for (int j = lane; j < S_; j += 32) {
        const float e = __expf(row[j] - m);
        row[j] = e; sum += e;
      }
#pragma unroll
      for (int o = 16; o > 0; o >>= 1) sum += __shfl_xor(sum, o, 32);
      const float inv = 1.f / sum;
      float* am = attn_mean + ((size_t)b * S_ + q0 + r) * S_;
      for (int j = lane; j < S_; j += 32) {
        const float p = row[j] * inv;
        row[j] = p;                               // P kept in LDS for P@V
        const float prev = (h == 0) ? 0.f : am[j];
        am[j] = prev + p * (1.f / H_);            // tile exclusively owned -> no atomics
      }
    }
    __syncthreads();

    // z tile: wave w owns features [w*16, w*16+16); P via ds_load_b128, V via 2x b128
    v8f accz = v8f_zero();
    for (int kc = 0; kc < S_; kc += 32) {
      const unsigned short* blkV = Vh + ((size_t)(kc >> 5) * 4 + wave) * 512;
      __builtin_prefetch(blkV + 2048, 0, 0);
      const v16bf ap = ld_a_f32(sc + kc, SSTR, 0);
      accz = wmma_bf16(ap, ld_b_swz(blkV), accz);
    }
    unsigned short* Zp = Zb + ((size_t)b * S_ + q0) * D_ + h * HD_ + wave * 16;
#pragma unroll
    for (int v = 0; v < 8; ++v)
      Zp[(size_t)(v + 8 * hs) * D_ + col] = f2bfu(accz[v]);
    __syncthreads();                              // sc reused next head
  }
}

// ---- kernel 4: out = Z @ Wz^T + bz (one wave = 16x64 tile, K = 1024) ----
__global__ __launch_bounds__(256) void out_proj(
    const unsigned short* __restrict__ Zb, const unsigned short* __restrict__ WzS,
    const float* __restrict__ bz, float* __restrict__ out) {
  const int lane = threadIdx.x & 31, wave = threadIdx.x >> 5;
  const int col = lane & 15, hs = lane >> 4;
  const int rt = blockIdx.x * 8 + wave;
  const int sg = rt * 16;                         // flattened (b*S + s) row
  const int n0 = blockIdx.y * 64;
  const unsigned short* Zp = Zb + (size_t)sg * D_;

  v8f acc[4] = {v8f_zero(), v8f_zero(), v8f_zero(), v8f_zero()};
  for (int kc = 0; kc < D_; kc += 32) {
    const v16bf a = ld_a_u16(Zp, D_, kc);
#pragma unroll
    for (int nt = 0; nt < 4; ++nt) {
      const unsigned short* blk = WzS + ((size_t)(kc >> 5) * 64 + (n0 >> 4) + nt) * 512;
      acc[nt] = wmma_bf16(a, ld_b_swz(blk), acc[nt]);
    }
  }
#pragma unroll
  for (int nt = 0; nt < 4; ++nt) {
    const float bias = bz[n0 + nt * 16 + col];
#pragma unroll
    for (int v = 0; v < 8; ++v)
      out[(size_t)(sg + v + 8 * hs) * D_ + n0 + nt * 16 + col] = acc[nt][v] + bias;
  }
}

extern "C" void kernel_launch(void* const* d_in, const int* in_sizes, int n_in,
                              void* d_out, int out_size, void* d_ws, size_t ws_size,
                              hipStream_t stream) {
  (void)in_sizes; (void)n_in; (void)out_size; (void)ws_size;
  const float* Xq = (const float*)d_in[0];
  const float* Xk = (const float*)d_in[1];
  const float* Xv = (const float*)d_in[2];
  const float* Wq = (const float*)d_in[3];
  const float* Wk = (const float*)d_in[4];
  const float* Wv = (const float*)d_in[5];
  const float* Wz = (const float*)d_in[6];
  const float* bz = (const float*)d_in[7];
  float* out       = (float*)d_out;
  float* attn_mean = out + (size_t)B_ * S_ * D_;

  unsigned short* ws = (unsigned short*)d_ws;
  size_t o = 0;
  unsigned short* WqS = ws + o; o += (size_t)H_ * HD_ * HD_;
  unsigned short* WkS = ws + o; o += (size_t)H_ * HD_ * HD_;
  unsigned short* WvS = ws + o; o += (size_t)H_ * HD_ * HD_;
  unsigned short* WzS = ws + o; o += (size_t)D_ * D_;
  unsigned short* Qb  = ws + o; o += (size_t)B_ * H_ * S_ * HD_;   // row-major
  unsigned short* KSw = ws + o; o += (size_t)B_ * H_ * S_ * HD_;   // Bt-swizzled
  unsigned short* VSw = ws + o; o += (size_t)B_ * H_ * S_ * HD_;   // B-swizzled
  unsigned short* Zb  = ws + o; o += (size_t)B_ * S_ * D_;         // row-major

  prep_weights<<<256, 256, 0, stream>>>(Wq, Wk, Wv, Wz, WqS, WkS, WvS, WzS);

  const float inv4 = 0.35355339059327373f;        // HD^-0.25, folded into Q and K
  dim3 pg(B_ * H_ * (S_ / 16) / 8);               // 512 blocks x 8 waves
  qkv_proj<<<pg, 256, 0, stream>>>(Xq, WqS, Qb,  inv4, 0);
  qkv_proj<<<pg, 256, 0, stream>>>(Xk, WkS, KSw, inv4, 2);
  qkv_proj<<<pg, 256, 0, stream>>>(Xv, WvS, VSw, 1.0f, 1);

  dim3 ag(S_ / 16, B_);                           // 128 x 2 workgroups
  attn_kernel<<<ag, 128, 16 * SSTR * sizeof(float), stream>>>(Qb, KSw, VSw, Zb, attn_mean);

  dim3 og((B_ * S_ / 16) / 8, D_ / 64);           // 32 x 16
  out_proj<<<og, 256, 0, stream>>>(Zb, WzS, bz, out);
}
